// MemoryEfficientAttention_29918742184323
// MI455X (gfx1250) — compile-verified, matrix operand order checked by reading, not run on hardware
//
#include <hip/hip_runtime.h>
#include <hip/hip_bf16.h>
#include <stdint.h>

typedef __attribute__((ext_vector_type(16))) _Float16 v16h;
typedef __attribute__((ext_vector_type(8)))  _Float16 v8h;
typedef __attribute__((ext_vector_type(8)))  float    v8f;

#define DIM    1024
#define NHEADS 16
#define HDIM   64
#define SEQ    4096
#define C3     3072          // 3 * DIM
#define ATT_SCALE 0.125f     // HDIM^-0.5

// Async copy of 16B global -> LDS (tracked by ASYNCcnt; no VGPR data).
__device__ __forceinline__ void async_g2l_b128(const _Float16* gptr, void* lptr) {
  uint32_t lds_off = (uint32_t)(uintptr_t)lptr;         // low 32b = LDS offset
  uint64_t gaddr   = (uint64_t)(uintptr_t)gptr;
  asm volatile("global_load_async_to_lds_b128 %0, %1, off"
               :: "v"(lds_off), "v"(gaddr) : "memory");
}
__device__ __forceinline__ void wait_async() {
  asm volatile("s_wait_asynccnt 0x0" ::: "memory");
}

// ---------------------------------------------------------------------------
// fp32 -> f16 conversion (one-time; halves GEMM load traffic, enables WMMA)
// ---------------------------------------------------------------------------
__global__ void f32_to_f16_kernel(const float* __restrict__ src,
                                  _Float16* __restrict__ dst, int n) {
  int i = blockIdx.x * blockDim.x + threadIdx.x;
  int stride = gridDim.x * blockDim.x;
  for (; i < n; i += stride) dst[i] = (_Float16)src[i];
}

// ---------------------------------------------------------------------------
// WMMA GEMM: Y[M,N] = X[M,K] @ W[N,K]^T  (+bias, fp32 out, when OUT_F32)
// 256 threads = 8 waves; block tile 128x128; K-step 32.
// Double-buffered LDS filled by GLOBAL_LOAD_ASYNC_TO_LDS_B128; one barrier
// + one s_wait_asynccnt per K-step; async fill of next tile overlaps WMMAs.
// ---------------------------------------------------------------------------
template <bool OUT_F32>
__global__ __launch_bounds__(256)
void wmma_gemm_kernel(const _Float16* __restrict__ X,
                      const _Float16* __restrict__ W,
                      const float* __restrict__ bias,
                      _Float16* __restrict__ Yh,
                      float* __restrict__ Yf,
                      int M, int N, int K) {
  __shared__ __align__(16) _Float16 Xs[2][128][32];
  __shared__ __align__(16) _Float16 Ws[2][128][32];

  const int tid   = threadIdx.x;
  const int lane  = tid & 31;
  const int wave  = tid >> 5;     // 0..7
  const int waveM = wave >> 2;    // 0..1 -> 64 rows
  const int waveN = wave & 3;     // 0..3 -> 32 cols
  const int lo    = lane & 15;
  const int hi    = lane >> 4;

  const int mBase = blockIdx.x * 128;
  const int nBase = blockIdx.y * 128;

  auto stage = [&](int kk, int buf) {
    for (int c = tid; c < 512; c += 256) {
      int m  = c >> 2;
      int k0 = (c & 3) << 3;
      async_g2l_b128(&X[(size_t)(mBase + m) * K + kk + k0], &Xs[buf][m][k0]);
      async_g2l_b128(&W[(size_t)(nBase + m) * K + kk + k0], &Ws[buf][m][k0]);
    }
  };

  v8f acc[4][2];
  for (int i = 0; i < 4; ++i)
    for (int j = 0; j < 2; ++j)
      for (int e = 0; e < 8; ++e) acc[i][j][e] = 0.0f;

  const int nk = K >> 5;
  stage(0, 0);  // prologue

  for (int t = 0; t < nk; ++t) {
    wait_async();       // our async fills of buf[t&1] are done
    __syncthreads();    // everyone's fills done / prior reads done
    const int cur = t & 1;
    if (t + 1 < nk) stage((t + 1) << 5, cur ^ 1);  // overlap with compute
    if (t + 2 < nk)  // warm L2 two tiles ahead
      __builtin_prefetch(&X[(size_t)(mBase + (tid >> 1)) * K + ((t + 2) << 5)], 0, 1);

    // fragments: per-lane K packing e0..7 = K(8hi+0..7), e8..15 = K(16+8hi+0..7)
    v16h a[4], b[2];
    for (int mt = 0; mt < 4; ++mt) {
      int ml = waveM * 64 + mt * 16 + lo;
      v8h p0 = *(const v8h*)&Xs[cur][ml][8 * hi];
      v8h p1 = *(const v8h*)&Xs[cur][ml][16 + 8 * hi];
      for (int e = 0; e < 8; ++e) { a[mt][e] = p0[e]; a[mt][e + 8] = p1[e]; }
    }
    for (int nt = 0; nt < 2; ++nt) {
      int nl = waveN * 32 + nt * 16 + lo;
      v8h p0 = *(const v8h*)&Ws[cur][nl][8 * hi];
      v8h p1 = *(const v8h*)&Ws[cur][nl][16 + 8 * hi];
      for (int e = 0; e < 8; ++e) { b[nt][e] = p0[e]; b[nt][e + 8] = p1[e]; }
    }

    for (int mt = 0; mt < 4; ++mt)
      for (int nt = 0; nt < 2; ++nt)
        acc[mt][nt] = __builtin_amdgcn_wmma_f32_16x16x32_f16(
            false, a[mt], false, b[nt], (short)0, acc[mt][nt], false, false);
  }

  // writeback: C/D layout -> M = r + 8*hi (VGPR r), N = lo
  for (int mt = 0; mt < 4; ++mt)
    for (int nt = 0; nt < 2; ++nt) {
      int row0 = mBase + waveM * 64 + mt * 16 + 8 * hi;
      int col  = nBase + waveN * 32 + nt * 16 + lo;
      if (OUT_F32) {
        float bv = bias ? bias[col] : 0.0f;
        for (int r = 0; r < 8; ++r)
          Yf[(size_t)(row0 + r) * N + col] = acc[mt][nt][r] + bv;
      } else {
        for (int r = 0; r < 8; ++r)
          Yh[(size_t)(row0 + r) * N + col] = (_Float16)acc[mt][nt][r];
      }
    }
}

// ---------------------------------------------------------------------------
// Flash attention (f16 WMMA, fp32 online softmax).
// Block = 128 threads (4 waves); block handles (head, 64-query tile); wave
// owns 16 queries. S^T = K@Q^T keeps the key-reduction in-lane (+1 shfl),
// and P^T's accumulator layout feeds B of O^T = V^T @ P^T element-wise.
// K tiles stream via async-to-LDS double buffering; V is transposed in-flight
// (vector global load + ds_store_b16 scatter).
// ---------------------------------------------------------------------------
__global__ __launch_bounds__(128)
void flash_attn_kernel(const _Float16* __restrict__ qkv,
                       _Float16* __restrict__ attn) {
  __shared__ __align__(16) _Float16 Kt[2][32][64];  // [key][d]
  __shared__ __align__(16) _Float16 Vt[2][64][48];  // [d][key], padded stride

  const int tid  = threadIdx.x;
  const int lane = tid & 31;
  const int wave = tid >> 5;  // 0..3
  const int lo   = lane & 15;
  const int hi   = lane >> 4;
  const int h    = blockIdx.y;
  const int q0   = blockIdx.x * 64 + wave * 16;

  const int qcol = h * HDIM;
  const int kcol = DIM + h * HDIM;
  const int vcol = 2 * DIM + h * HDIM;

  auto stageK = [&](int j, int buf) {  // async global -> LDS
    for (int c = tid; c < 256; c += 128) {
      int key = c >> 3;
      int d0  = (c & 7) << 3;
      async_g2l_b128(&qkv[(size_t)(j + key) * C3 + kcol + d0],
                     &Kt[buf][key][d0]);
    }
  };
  auto stageV = [&](int j, int buf) {  // transpose on the way into LDS
    for (int c = tid; c < 256; c += 128) {
      int key = c >> 3;
      int d0  = (c & 7) << 3;
      v8h vv = *(const v8h*)&qkv[(size_t)(j + key) * C3 + vcol + d0];
      for (int e = 0; e < 8; ++e) Vt[buf][d0 + e][key] = vv[e];
    }
  };

  // Q fragments (B operand of S^T), two K-steps of 32 covering hd=64
  v16h qb[2];
  {
    const size_t base = (size_t)(q0 + lo) * C3 + qcol;
    for (int s = 0; s < 2; ++s) {
      v8h p0 = *(const v8h*)&qkv[base + s * 32 + 8 * hi];
      v8h p1 = *(const v8h*)&qkv[base + s * 32 + 16 + 8 * hi];
      for (int e = 0; e < 8; ++e) { qb[s][e] = p0[e]; qb[s][e + 8] = p1[e]; }
    }
  }

  v8f accO[4];  // O^T accum: lane holds (d = t*16 + r + 8*hi, q = lo)
  for (int t = 0; t < 4; ++t)
    for (int e = 0; e < 8; ++e) accO[t][e] = 0.0f;
  float m_run = -1e30f, l_run = 0.0f;

  const int njt = SEQ >> 5;
  stageK(0, 0);
  stageV(0, 0);

  for (int jt = 0; jt < njt; ++jt) {
    wait_async();
    __syncthreads();
    const int cur = jt & 1;
    if (jt + 1 < njt) {  // overlap next tile's fill with this tile's math
      stageK((jt + 1) << 5, cur ^ 1);
      stageV((jt + 1) << 5, cur ^ 1);
    }

    // S^T = K @ Q^T : 2 key halves x 2 hd-steps = 4 WMMAs
    v8f accS[2];
    for (int a = 0; a < 2; ++a)
      for (int e = 0; e < 8; ++e) accS[a][e] = 0.0f;
    for (int a = 0; a < 2; ++a) {
      int keyl = a * 16 + lo;
      for (int s = 0; s < 2; ++s) {
        v16h ka;
        v8h p0 = *(const v8h*)&Kt[cur][keyl][s * 32 + 8 * hi];
        v8h p1 = *(const v8h*)&Kt[cur][keyl][s * 32 + 16 + 8 * hi];
        for (int e = 0; e < 8; ++e) { ka[e] = p0[e]; ka[e + 8] = p1[e]; }
        accS[a] = __builtin_amdgcn_wmma_f32_16x16x32_f16(
            false, ka, false, qb[s], (short)0, accS[a], false, false);
      }
    }

    // online softmax (per query column q = lo)
    float tmax = -1e30f;
    for (int a = 0; a < 2; ++a)
      for (int e = 0; e < 8; ++e) {
        float sc = accS[a][e] * ATT_SCALE;
        accS[a][e] = sc;
        tmax = fmaxf(tmax, sc);
      }
    tmax = fmaxf(tmax, __shfl_xor(tmax, 16, 32));
    float m_new = fmaxf(m_run, tmax);
    float corr  = __expf(m_run - m_new);
    float psum  = 0.0f;
    v16h pb;  // B fragment of P^T: e0..7 <- keys 8hi+r, e8..15 <- keys 16+8hi+r
    for (int a = 0; a < 2; ++a)
      for (int e = 0; e < 8; ++e) {
        float p = __expf(accS[a][e] - m_new);
        psum += p;
        pb[a * 8 + e] = (_Float16)p;
      }
    psum += __shfl_xor(psum, 16, 32);
    l_run = l_run * corr + psum;
    m_run = m_new;
    for (int t = 0; t < 4; ++t)
      for (int e = 0; e < 8; ++e) accO[t][e] *= corr;

    // O^T += V^T @ P^T : 4 d-tiles
    for (int t = 0; t < 4; ++t) {
      v16h va;
      int dl = t * 16 + lo;
      v8h p0 = *(const v8h*)&Vt[cur][dl][8 * hi];
      v8h p1 = *(const v8h*)&Vt[cur][dl][16 + 8 * hi];
      for (int e = 0; e < 8; ++e) { va[e] = p0[e]; va[e + 8] = p1[e]; }
      accO[t] = __builtin_amdgcn_wmma_f32_16x16x32_f16(
          false, va, false, pb, (short)0, accO[t], false, false);
    }
  }

  // normalize and store O (coalesced 16B f16 chunks)
  float inv = 1.0f / l_run;
  for (int t = 0; t < 4; ++t) {
    v8h o;
    for (int e = 0; e < 8; ++e) o[e] = (_Float16)(accO[t][e] * inv);
    int n   = q0 + lo;
    int col = h * HDIM + t * 16 + 8 * hi;
    *(v8h*)&attn[(size_t)n * DIM + col] = o;
  }
}

// ---------------------------------------------------------------------------
extern "C" void kernel_launch(void* const* d_in, const int* in_sizes, int n_in,
                              void* d_out, int out_size, void* d_ws,
                              size_t ws_size, hipStream_t stream) {
  (void)in_sizes; (void)n_in; (void)out_size; (void)ws_size;
  const float* x     = (const float*)d_in[0];
  const float* Wqkv  = (const float*)d_in[1];
  const float* Wproj = (const float*)d_in[2];
  const float* bproj = (const float*)d_in[3];
  float* out = (float*)d_out;

  char* ws = (char*)d_ws;
  size_t off = 0;
  auto alloc = [&](size_t bytes) {
    void* p = ws + off;
    off = (off + bytes + 255) & ~(size_t)255;
    return p;
  };
  _Float16* xh     = (_Float16*)alloc((size_t)SEQ * C3 * 2);
  _Float16* wqkvh  = (_Float16*)alloc((size_t)C3 * C3 * 2);
  _Float16* wprojh = (_Float16*)alloc((size_t)DIM * DIM * 2);
  _Float16* qkvh   = (_Float16*)alloc((size_t)SEQ * C3 * 2);
  _Float16* attnh  = (_Float16*)alloc((size_t)SEQ * DIM * 2);

  // 1) fp32 -> f16 conversions
  f32_to_f16_kernel<<<2048, 256, 0, stream>>>(x, xh, SEQ * C3);
  f32_to_f16_kernel<<<2048, 256, 0, stream>>>(Wqkv, wqkvh, C3 * C3);
  f32_to_f16_kernel<<<512, 256, 0, stream>>>(Wproj, wprojh, DIM * DIM);

  // 2) QKV projection: qkv = x @ Wqkv^T   [4096, 3072]
  wmma_gemm_kernel<false><<<dim3(SEQ / 128, C3 / 128), 256, 0, stream>>>(
      xh, wqkvh, nullptr, qkvh, nullptr, SEQ, C3, C3);

  // 3) flash attention per (query-tile, head)
  flash_attn_kernel<<<dim3(SEQ / 64, NHEADS), 128, 0, stream>>>(qkvh, attnh);

  // 4) output projection: out = attn @ Wproj^T + b   (fp32 out)
  wmma_gemm_kernel<true><<<dim3(SEQ / 128, DIM / 128), 256, 0, stream>>>(
      attnh, wprojh, bproj, nullptr, out, SEQ, DIM, DIM);
}
